// EventClassifierTransformerNetwork_38285338476741
// MI455X (gfx1250) — matrix operand order repaired and checked by reading, not compiled, not run on hardware
//
#include <hip/hip_runtime.h>
#include <hip/hip_bf16.h>

#define NVAR 32
#define BATCH 32768
#define EDIM 16
#define NHEAD 4
#define HDIM 4
#define FFDIM 64

typedef __attribute__((ext_vector_type(16))) _Float16 v16h;
typedef __attribute__((ext_vector_type(8)))  _Float16 v8h;
typedef __attribute__((ext_vector_type(8)))  float    v8f;

__device__ __forceinline__ float gelu_f(float x) {
    return 0.5f * x * (1.0f + erff(x * 0.70710678118654752f));
}

__device__ __forceinline__ float wave_max32(float v) {
#pragma unroll
    for (int m = 16; m >= 1; m >>= 1) v = fmaxf(v, __shfl_xor(v, m, 32));
    return v;
}
__device__ __forceinline__ float wave_sum32(float v) {
#pragma unroll
    for (int m = 16; m >= 1; m >>= 1) v += __shfl_xor(v, m, 32);
    return v;
}
__device__ __forceinline__ float grp16_sum(float v) {
#pragma unroll
    for (int m = 8; m >= 1; m >>= 1) v += __shfl_xor(v, m, 32);
    return v;
}

// ---- WMMA helpers (D = A * B + C, 16x16x32 f16 -> f32) ----------------------

__device__ __forceinline__ v8f wmma16(v16h a, v16h b, v8f c) {
    return __builtin_amdgcn_wmma_f32_16x16x32_f16(
        /*neg_a=*/false, a, /*neg_b=*/false, b,
        /*c_mod=*/(short)0, c, /*reuse_a=*/false, /*reuse_b=*/false);
}

// A fragment (16x32 f16): lanes 0-15 row M=lane, K0..15 (el 0..15)
//                         lanes 16-31 row M=lane-16, K16..31
__device__ __forceinline__ v16h ldsA(const _Float16* base, int stride, int mt, int kc) {
    const int l    = threadIdx.x & 31;
    const int hsel = l >> 4;
    const int row  = mt * 16 + (l & 15);
    const _Float16* p = base + row * stride + kc + hsel * 16;
    v8h lo = *(const v8h*)(p);
    v8h hi = *(const v8h*)(p + 8);
    v16h a;
#pragma unroll
    for (int i = 0; i < 8; ++i) { a[i] = lo[i]; a[8 + i] = hi[i]; }
    return a;
}

// B fragment (32x16 f16): lane L<16 holds column n=nrow0+L, elements = K 0..15;
// lanes 16-31 hold K 16..31 of that column. GEMM is X @ W^T so B[k][n] = W[n][k].
// Real K = 16 (padded): upper-half lanes are exactly the zero pad.
__device__ __forceinline__ v16h ldgB_k16(const float* W, int ldw, int nrow0) {
    const int l = threadIdx.x & 31;
    const float* p = W + (nrow0 + (l & 15)) * ldw;
    const bool lo = (l < 16);
    v16h b;
#pragma unroll
    for (int i = 0; i < 16; ++i) b[i] = lo ? (_Float16)p[i] : (_Float16)0.0f;
    return b;
}

// Real K = 32 chunk (for K=64 GEMMs): kc in {0,32}
__device__ __forceinline__ v16h ldgB_k32(const float* W, int ldw, int nrow0, int kc) {
    const int l = threadIdx.x & 31;
    const float* p = W + (nrow0 + (l & 15)) * ldw + kc + ((l >> 4) ? 16 : 0);
    v16h b;
#pragma unroll
    for (int i = 0; i < 16; ++i) b[i] = (_Float16)p[i];
    return b;
}

// C/D layout: VGPR r -> row mt*16 + r (lanes 0-15) / mt*16 + 8 + r (lanes 16-31), col = lane&15
__device__ __forceinline__ void storeC(const v8f& c, _Float16* base, int stride,
                                       int mt, int ncol0, int rowLimit) {
    const int l    = threadIdx.x & 31;
    const int col  = ncol0 + (l & 15);
    const int row0 = mt * 16 + (l >> 4) * 8;
#pragma unroll
    for (int r = 0; r < 8; ++r) {
        int row = row0 + r;
        if (row < rowLimit) base[row * stride + col] = (_Float16)c[r];
    }
}

// store with bias + residual-add (residual read per element from resBase)
__device__ __forceinline__ void storeC_addres(const v8f& c, float biasv,
                                              const _Float16* resBase, int resStride,
                                              _Float16* dst, int dstStride,
                                              int mt, int col0, int rowOff) {
    const int l    = threadIdx.x & 31;
    const int col  = col0 + (l & 15);
    const int row0 = mt * 16 + (l >> 4) * 8;
#pragma unroll
    for (int r = 0; r < 8; ++r) {
        int row = row0 + r;
        float v = c[r] + biasv + (float)resBase[row * resStride + col];
        dst[(row + rowOff) * dstStride + col] = (_Float16)v;
    }
}

// per-row LayerNorm over 16 cols (lane-per-row)
__device__ __forceinline__ void ln_row16(const _Float16* src, _Float16* dst,
                                         const float* g, const float* bb) {
    float v[16]; float s = 0.f;
#pragma unroll
    for (int i = 0; i < 16; ++i) { v[i] = (float)src[i]; s += v[i]; }
    float m = s * (1.0f / 16.0f);
    float vs = 0.f;
#pragma unroll
    for (int i = 0; i < 16; ++i) { float d = v[i] - m; vs += d * d; }
    float inv = rsqrtf(vs * (1.0f / 16.0f) + 1e-5f);
#pragma unroll
    for (int i = 0; i < 16; ++i)
        dst[i] = (_Float16)((v[i] - m) * inv * g[i] + bb[i]);
}

// ---- fused network kernel: one wave per batch element ----------------------
// LDS per wave: B0 [48 x 80] f16 (wide activations / u), B1 [33 x 40], B2 [33 x 40]

#define S0 80
#define S1 40
#define WAVE_HALVES (48 * S0 + 2 * (33 * S1))   // 6480 halves = 12960 B
#define WPB 4

__global__ __launch_bounds__(128)
void EventClassifierTransformerNetwork_38285338476741_kernel(
    const float* __restrict__ x,
    const float* __restrict__ emb_w1, const float* __restrict__ emb_b1,
    const float* __restrict__ emb_w2, const float* __restrict__ emb_b2,
    const float* __restrict__ ln1_g,  const float* __restrict__ ln1_b,
    const float* __restrict__ attn_wi, const float* __restrict__ attn_bi,
    const float* __restrict__ attn_wo, const float* __restrict__ attn_bo,
    const float* __restrict__ ln2_g,  const float* __restrict__ ln2_b,
    const float* __restrict__ fc1_w,  const float* __restrict__ fc1_b,
    const float* __restrict__ fc2_w,  const float* __restrict__ fc2_b,
    const float* __restrict__ cls_token,
    const float* __restrict__ cln1_g, const float* __restrict__ cln1_b,
    const float* __restrict__ cattn_wi, const float* __restrict__ cattn_bi,
    const float* __restrict__ cattn_wo, const float* __restrict__ cattn_bo,
    const float* __restrict__ cln2_g, const float* __restrict__ cln2_b,
    const float* __restrict__ clsfc_w, const float* __restrict__ clsfc_b,
    float* __restrict__ out)
{
    __shared__ _Float16 smem[WPB * WAVE_HALVES];
    const int wid = threadIdx.x >> 5;
    const int l   = threadIdx.x & 31;
    _Float16* B0 = smem + wid * WAVE_HALVES;     // 48 rows x 80
    _Float16* B1 = B0 + 48 * S0;                 // 33 rows x 40
    _Float16* B2 = B1 + 33 * S1;                 // 33 rows x 40
    const int b = blockIdx.x * WPB + wid;

    __builtin_prefetch(attn_wi, 0, 3);
    __builtin_prefetch(fc1_w, 0, 3);
    __builtin_prefetch(fc2_w, 0, 3);

    // zero this wave's LDS region (also provides the WMMA K-padding zeros)
    {
        uint4 z; z.x = 0; z.y = 0; z.z = 0; z.w = 0;
        uint4* p = (uint4*)B0;
        const int n16 = WAVE_HALVES * 2 / 16;
        for (int i = l; i < n16; i += 32) p[i] = z;
    }

    // ---- per-feature tokenizer (lane = variable v) -> T in B1 rows 0..31 ----
    {
        float xv = x[b * NVAR + l];
        float hh[4];
#pragma unroll
        for (int f = 0; f < 4; ++f)
            hh[f] = gelu_f(xv * emb_w1[l * 4 + f] + emb_b1[l * 4 + f]);
        __attribute__((aligned(16))) _Float16 row[16];
#pragma unroll
        for (int e = 0; e < 16; ++e) {
            float acc = emb_b2[l * 16 + e];
#pragma unroll
            for (int f = 0; f < 4; ++f) acc += hh[f] * emb_w2[(l * 4 + f) * 16 + e];
            row[e] = (_Float16)gelu_f(acc);
        }
        v8h* d = (v8h*)(B1 + l * S1);
        d[0] = *(v8h*)row; d[1] = *(v8h*)(row + 8);
    }

    // ---- LN1: B1 -> B2 ----
    ln_row16(B1 + l * S1, B2 + l * S1, ln1_g, ln1_b);

    // ---- QKV projection: B0 cols [0:16)=Q, [16:32)=K, [32:48)=V  (6 WMMA) ----
#pragma unroll
    for (int nt = 0; nt < 3; ++nt) {
        v16h bf = ldgB_k16(attn_wi, EDIM, nt * 16);
        float biasv = attn_bi[nt * 16 + (l & 15)];
#pragma unroll
        for (int mt = 0; mt < 2; ++mt) {
            v16h a = ldsA(B2, S1, mt, 0);
            v8f c = {};
            c = wmma16(a, bf, c);
            v8f cc;
#pragma unroll
            for (int r = 0; r < 8; ++r) cc[r] = c[r] + biasv;
            storeC(cc, B0, S0, mt, nt * 16, 48);
        }
    }

    // ---- self-attention core (lane = key, wave32 butterfly softmax) -> B2 ----
    for (int q = 0; q < 32; ++q) {
#pragma unroll
        for (int h = 0; h < NHEAD; ++h) {
            float s = 0.f;
#pragma unroll
            for (int d = 0; d < HDIM; ++d)
                s += (float)B0[q * S0 + h * 4 + d] * (float)B0[l * S0 + 16 + h * 4 + d];
            s *= 0.5f;  // 1/sqrt(D)
            float m = wave_max32(s);
            float p = __expf(s - m);
            float Z = wave_sum32(p);
            float o[4];
#pragma unroll
            for (int d = 0; d < HDIM; ++d)
                o[d] = wave_sum32(p * (float)B0[l * S0 + 32 + h * 4 + d]) / Z;
            if (l == 0) {
#pragma unroll
                for (int d = 0; d < HDIM; ++d)
                    B2[q * S1 + h * 4 + d] = (_Float16)o[d];
            }
        }
    }

    // ---- O projection + residual -> B1 (2 WMMA) ----
    {
        v16h bf = ldgB_k16(attn_wo, EDIM, 0);
        float biasv = attn_bo[l & 15];
#pragma unroll
        for (int mt = 0; mt < 2; ++mt) {
            v16h a = ldsA(B2, S1, mt, 0);
            v8f c = {};
            c = wmma16(a, bf, c);
            storeC_addres(c, biasv, B1, S1, B1, S1, mt, 0, 0);
        }
    }

    // ---- LN2: B1 -> B2 ----
    ln_row16(B1 + l * S1, B2 + l * S1, ln2_g, ln2_b);

    // ---- FC1 (16 -> 64) + GELU -> B0 cols 0..63 (8 WMMA) ----
#pragma unroll
    for (int nt = 0; nt < 4; ++nt) {
        v16h bf = ldgB_k16(fc1_w, EDIM, nt * 16);
        float biasv = fc1_b[nt * 16 + (l & 15)];
#pragma unroll
        for (int mt = 0; mt < 2; ++mt) {
            v16h a = ldsA(B2, S1, mt, 0);
            v8f c = {};
            c = wmma16(a, bf, c);
            v8f cg;
#pragma unroll
            for (int r = 0; r < 8; ++r) cg[r] = gelu_f(c[r] + biasv);
            storeC(cg, B0, S0, mt, nt * 16, 48);
        }
    }

    // ---- FC2 (64 -> 16, K=64 in two chunks) + residual -> u rows 1..32 (4 WMMA) ----
    {
        float biasv = fc2_b[l & 15];
        v16h bk0 = ldgB_k32(fc2_w, FFDIM, 0, 0);
        v16h bk1 = ldgB_k32(fc2_w, FFDIM, 0, 32);
        v16h a00 = ldsA(B0, S0, 0, 0);
        v16h a01 = ldsA(B0, S0, 0, 32);
        v16h a10 = ldsA(B0, S0, 1, 0);
        v16h a11 = ldsA(B0, S0, 1, 32);
        v8f c0 = {}; c0 = wmma16(a00, bk0, c0); c0 = wmma16(a01, bk1, c0);
        v8f c1 = {}; c1 = wmma16(a10, bk0, c1); c1 = wmma16(a11, bk1, c1);
        storeC_addres(c0, biasv, B1, S1, B0, S0, 0, 0, 1);
        storeC_addres(c1, biasv, B1, S1, B0, S0, 1, 0, 1);
    }

    // ---- build u in B0: row 0 = cls token; zero the K-pad cols 16..31 ----
    {
        v8h z = {};
        for (int rr = l; rr < 48; rr += 32) {
            *(v8h*)(B0 + rr * S0 + 16) = z;
            *(v8h*)(B0 + rr * S0 + 24) = z;
        }
        if (l < 16) B0[l] = (_Float16)cls_token[l];
    }

    // ---- cln1 over u rows 0..32 (in place) ----
#pragma unroll
    for (int r0 = 0; r0 < 64; r0 += 32) {
        int row = r0 + l;
        if (row < 33) ln_row16(B0 + row * S0, B0 + row * S0, cln1_g, cln1_b);
    }

    // ---- cross-attn K/V projection over 48-padded u rows (6 WMMA) ----
    {
        v16h bk = ldgB_k16(cattn_wi + 16 * EDIM, EDIM, 0);
        v16h bv = ldgB_k16(cattn_wi + 32 * EDIM, EDIM, 0);
        float bkb = cattn_bi[16 + (l & 15)];
        float bvb = cattn_bi[32 + (l & 15)];
#pragma unroll
        for (int mt = 0; mt < 3; ++mt) {
            v16h a = ldsA(B0, S0, mt, 0);
            v8f ck = {}; ck = wmma16(a, bk, ck);
            v8f cv = {}; cv = wmma16(a, bv, cv);
            v8f tk, tv;
#pragma unroll
            for (int r = 0; r < 8; ++r) { tk[r] = ck[r] + bkb; tv[r] = cv[r] + bvb; }
            storeC(tk, B1, S1, mt, 0, 33);   // K2 -> B1 cols 0..15, rows 0..32
            storeC(tv, B1, S1, mt, 16, 33);  // V2 -> B1 cols 16..31
        }
    }

    // ---- query projection of raw cls token -> B2 row 0 ----
    if (l < 16) {
        float acc = cattn_bi[l];
#pragma unroll
        for (int j = 0; j < 16; ++j) acc += cattn_wi[l * EDIM + j] * cls_token[j];
        B2[l] = (_Float16)acc;
    }

    // ---- cross attention: 1 query, 33 keys (lane = key 0..31; key 32 scalar) ----
#pragma unroll
    for (int h = 0; h < NHEAD; ++h) {
        float qv[4];
#pragma unroll
        for (int d = 0; d < HDIM; ++d) qv[d] = (float)B2[h * 4 + d];
        float s = 0.f, s32 = 0.f;
#pragma unroll
        for (int d = 0; d < HDIM; ++d) {
            s   += qv[d] * (float)B1[l  * S1 + h * 4 + d];
            s32 += qv[d] * (float)B1[32 * S1 + h * 4 + d];
        }
        s *= 0.5f; s32 *= 0.5f;
        float m = fmaxf(wave_max32(s), s32);
        float p = __expf(s - m), p32 = __expf(s32 - m);
        float Z = wave_sum32(p) + p32;
#pragma unroll
        for (int d = 0; d < HDIM; ++d) {
            float o = (wave_sum32(p * (float)B1[l * S1 + 16 + h * 4 + d])
                       + p32 * (float)B1[32 * S1 + 16 + h * 4 + d]) / Z;
            if (l == 0) B2[S1 + h * 4 + d] = (_Float16)o;
        }
    }

    // ---- cross O-proj + cls residual, cln2 (16-lane group LN), classifier ----
    float cvv = 0.f;
    if (l < 16) {
        float acc = cattn_bo[l] + cls_token[l];
#pragma unroll
        for (int j = 0; j < 16; ++j) acc += cattn_wo[l * EDIM + j] * (float)B2[S1 + j];
        cvv = acc;
    }
    float mean = grp16_sum(cvv) * (1.0f / 16.0f);
    float dd = cvv - mean;
    float var = grp16_sum((l < 16) ? dd * dd : 0.0f) * (1.0f / 16.0f);
    float inv = rsqrtf(var + 1e-5f);
    float y = (l < 16) ? (dd * inv * cln2_g[l] + cln2_b[l]) : 0.f;
    float zsum = grp16_sum((l < 16) ? y * clsfc_w[l] : 0.f);
    if (l == 0) out[b] = 1.f / (1.f + __expf(-(zsum + clsfc_b[0])));
}

extern "C" void kernel_launch(void* const* d_in, const int* in_sizes, int n_in,
                              void* d_out, int out_size, void* d_ws, size_t ws_size,
                              hipStream_t stream) {
    const float* x        = (const float*)d_in[0];
    const float* emb_w1   = (const float*)d_in[1];
    const float* emb_b1   = (const float*)d_in[2];
    const float* emb_w2   = (const float*)d_in[3];
    const float* emb_b2   = (const float*)d_in[4];
    const float* ln1_g    = (const float*)d_in[5];
    const float* ln1_b    = (const float*)d_in[6];
    const float* attn_wi  = (const float*)d_in[7];
    const float* attn_bi  = (const float*)d_in[8];
    const float* attn_wo  = (const float*)d_in[9];
    const float* attn_bo  = (const float*)d_in[10];
    const float* ln2_g    = (const float*)d_in[11];
    const float* ln2_b    = (const float*)d_in[12];
    const float* fc1_w    = (const float*)d_in[13];
    const float* fc1_b    = (const float*)d_in[14];
    const float* fc2_w    = (const float*)d_in[15];
    const float* fc2_b    = (const float*)d_in[16];
    const float* cls_tok  = (const float*)d_in[17];
    const float* cln1_g   = (const float*)d_in[18];
    const float* cln1_b   = (const float*)d_in[19];
    const float* cattn_wi = (const float*)d_in[20];
    const float* cattn_bi = (const float*)d_in[21];
    const float* cattn_wo = (const float*)d_in[22];
    const float* cattn_bo = (const float*)d_in[23];
    const float* cln2_g   = (const float*)d_in[24];
    const float* cln2_b   = (const float*)d_in[25];
    const float* clsfc_w  = (const float*)d_in[26];
    const float* clsfc_b  = (const float*)d_in[27];
    float* out = (float*)d_out;

    dim3 grid(BATCH / WPB), block(WPB * 32);
    EventClassifierTransformerNetwork_38285338476741_kernel<<<grid, block, 0, stream>>>(
        x, emb_w1, emb_b1, emb_w2, emb_b2, ln1_g, ln1_b,
        attn_wi, attn_bi, attn_wo, attn_bo, ln2_g, ln2_b,
        fc1_w, fc1_b, fc2_w, fc2_b, cls_tok,
        cln1_g, cln1_b, cattn_wi, cattn_bi, cattn_wo, cattn_bo,
        cln2_g, cln2_b, clsfc_w, clsfc_b, out);
}
